// FusionModel_31602369364760
// MI455X (gfx1250) — matrix operand order repaired and checked by reading, not compiled, not run on hardware
//
#include <hip/hip_runtime.h>

typedef __attribute__((ext_vector_type(16))) __bf16 v16bf;
typedef __attribute__((ext_vector_type(8)))  __bf16 v8bf;
typedef __attribute__((ext_vector_type(8)))  float  v8f;
typedef __attribute__((ext_vector_type(4)))  float  v4f;
typedef __attribute__((ext_vector_type(4)))  int    v4i;

// Model dims
#define BATCH 4096
#define DIMA  768
#define DIMV  2048
#define DIMH  512
#define DIMF  1024
#define ND    16
#define NC    2
#define NE    8

// GEMM tiling
#define BM 64
#define BN 128
#define BK 32
#define RS (BK + 8)   // LDS row stride in bf16 elems: 40 elems = 80 bytes (16B aligned, bank-spread)

// CDNA5 async global->LDS path (guarded: falls back to sync staging if the
// toolchain does not declare the gfx1250 async builtins)
#if defined(__has_builtin)
#if __has_builtin(__builtin_amdgcn_global_load_async_to_lds_b128) && \
    __has_builtin(__builtin_amdgcn_s_wait_asynccnt)
#define USE_ASYNC_LDS 1
#endif
#endif
#ifndef USE_ASYNC_LDS
#define USE_ASYNC_LDS 0
#endif

// builtin expects typed int4 pointers: global (AS1) source, LDS (AS3) dest
typedef __attribute__((address_space(1))) v4i g_v4i;
typedef __attribute__((address_space(3))) v4i l_v4i;

// generic->LDS: hardware maps flat->LDS by truncating to addr[31:0]
__device__ __forceinline__ l_v4i* as_lds_v4i(void* p) {
    return (l_v4i*)(unsigned)(unsigned long long)p;
}
__device__ __forceinline__ g_v4i* as_gbl_v4i(const void* p) {
    return (g_v4i*)(unsigned long long)p;
}

// ---------------------------------------------------------------------------
// fp32 (K x N) -> bf16 transposed (N x K), batched over gridDim.z
// ---------------------------------------------------------------------------
__global__ __launch_bounds__(256)
void k_transpose_bf16(const float* __restrict__ W, __bf16* __restrict__ WT,
                      int K, int N, long inBatchStride, long outBatchStride) {
    __shared__ float tile[32][33];
    const float* Wb  = W  + (long)blockIdx.z * inBatchStride;
    __bf16*      WTb = WT + (long)blockIdx.z * outBatchStride;
    const int n0 = blockIdx.x * 32;
    const int k0 = blockIdx.y * 32;
    {
        const int c  = threadIdx.x & 31;
        const int r4 = (threadIdx.x >> 5) * 4;
#pragma unroll
        for (int j = 0; j < 4; ++j)
            tile[r4 + j][c] = Wb[(long)(k0 + r4 + j) * N + n0 + c];
    }
    __syncthreads();
    {
        const int kk = threadIdx.x & 31;
        const int n4 = (threadIdx.x >> 5) * 4;
#pragma unroll
        for (int j = 0; j < 4; ++j)
            WTb[(long)(n0 + n4 + j) * K + k0 + kk] = (__bf16)tile[kk][n4 + j];
    }
}

// ---------------------------------------------------------------------------
// out(MxN) = act( A1(MxK1) @ W1 + [A2(MxK2) @ W2] + bias ),  W given as
// pre-transposed bf16 (N x K).  A is fp32, converted to bf16 while staging.
// Block = 256 thr = 8 wave32, tile 64x128, each wave does 32x32 via 4 WMMAs.
// W tiles stream through GLOBAL_LOAD_ASYNC_TO_LDS_B128 (ASYNCcnt) when the
// toolchain exposes it; A tiles use the VALU cvt path they need anyway.
// ---------------------------------------------------------------------------
__device__ __forceinline__ v8bf cvt8_f32_bf16(const float* __restrict__ s) {
    v4f f0 = *(const v4f*)(s);
    v4f f1 = *(const v4f*)(s + 4);
    v8bf r;
#pragma unroll
    for (int i = 0; i < 4; ++i) { r[i] = (__bf16)f0[i]; r[4 + i] = (__bf16)f1[i]; }
    return r;
}

__global__ __launch_bounds__(256)
void k_gemm_bf16(const float* __restrict__ A1, const __bf16* __restrict__ W1, int K1,
                 const float* __restrict__ A2, const __bf16* __restrict__ W2, int K2,
                 const float* __restrict__ bias, float* __restrict__ out,
                 int N, int doRelu) {
    __shared__ __bf16 sA[BM * RS];
    __shared__ __bf16 sB[BN * RS];

    const int t    = threadIdx.x;
    const int lane = t & 31;
    const int wave = t >> 5;
    const int wm   = wave & 1;   // 0..1  -> M offset 0/32
    const int wn   = wave >> 1;  // 0..3  -> N offset 0/32/64/96
    const int m0   = blockIdx.x * BM;
    const int n0   = blockIdx.y * BN;
    const int half = lane >> 4;  // 0: lanes 0-15, 1: lanes 16-31
    const int lm   = lane & 15;

    v8f acc[2][2] = {};

#pragma unroll 1
    for (int pair = 0; pair < 2; ++pair) {
        const float*  A = pair ? A2 : A1;
        const __bf16* W = pair ? W2 : W1;
        const int     K = pair ? K2 : K1;
        if (K == 0) continue;

        for (int k0 = 0; k0 < K; k0 += BK) {
            // ---- stage A tile (BM x BK) fp32 -> bf16, row-major in LDS ----
            {
                const int row = t >> 2;
                const int kc  = (t & 3) * 8;
                const float* src = A + (long)(m0 + row) * K + k0 + kc;
                __builtin_prefetch(src + BK, 0, 1);   // speculative; dropped if OOB
                *(v8bf*)(&sA[row * RS + kc]) = cvt8_f32_bf16(src);
            }
            // ---- stage W tile (BN x BK) from N-major bf16 weights ----
            {
                const int n  = t >> 1;
                const int kc = (t & 1) * 16;
                const __bf16* src = W + (long)(n0 + n) * K + k0 + kc;
                __bf16*       dst = &sB[n * RS + kc];
#if USE_ASYNC_LDS
                __builtin_amdgcn_global_load_async_to_lds_b128(as_gbl_v4i(src),     as_lds_v4i(dst),     0, 0);
                __builtin_amdgcn_global_load_async_to_lds_b128(as_gbl_v4i(src + 8), as_lds_v4i(dst + 8), 0, 0);
#else
                *(v8bf*)(dst)     = *(const v8bf*)(src);
                *(v8bf*)(dst + 8) = *(const v8bf*)(src + 8);
#endif
            }
#if USE_ASYNC_LDS
            __builtin_amdgcn_s_wait_asynccnt(0);
#endif
            __syncthreads();

            // ---- build fragments per ISA 16-bit A(16x32)/B(32x16) layouts ----
            v16bf afrag[2], bfrag[2];
#pragma unroll
            for (int sm = 0; sm < 2; ++sm) {
                const int row = wm * 32 + sm * 16 + lm;
                const int c0  = half * 8;                     // K 0..7 | 8..15
                v8bf lo = *(const v8bf*)(&sA[row * RS + c0]);
                v8bf hi = *(const v8bf*)(&sA[row * RS + c0 + 16]); // +K16
                afrag[sm] = __builtin_shufflevector(lo, hi,
                    0,1,2,3,4,5,6,7,8,9,10,11,12,13,14,15);
            }
#pragma unroll
            for (int sn = 0; sn < 2; ++sn) {
                const int row = wn * 32 + sn * 16 + lm;       // row = n in sB
                const int c0  = half * 8;                     // K 0..7 | 8..15
                v8bf lo = *(const v8bf*)(&sB[row * RS + c0]);
                v8bf hi = *(const v8bf*)(&sB[row * RS + c0 + 16]); // +K16
                bfrag[sn] = __builtin_shufflevector(lo, hi,
                    0,1,2,3,4,5,6,7,8,9,10,11,12,13,14,15);
            }
#pragma unroll
            for (int sm = 0; sm < 2; ++sm)
#pragma unroll
                for (int sn = 0; sn < 2; ++sn)
                    acc[sm][sn] = __builtin_amdgcn_wmma_f32_16x16x32_bf16(
                        false, afrag[sm], false, bfrag[sn],
                        (short)0, acc[sm][sn], false, false);
            __syncthreads();
        }
    }

    // ---- epilogue: bias + relu, D layout: VGPR r -> M = r + half*8, N = lm ----
#pragma unroll
    for (int sm = 0; sm < 2; ++sm)
#pragma unroll
        for (int sn = 0; sn < 2; ++sn) {
            const int n = n0 + wn * 32 + sn * 16 + lm;
            const float bv = bias ? bias[n] : 0.0f;
#pragma unroll
            for (int r = 0; r < 8; ++r) {
                const int m = m0 + wm * 32 + sm * 16 + half * 8 + r;
                float v = acc[sm][sn][r] + bv;
                if (doRelu) v = fmaxf(v, 0.0f);
                out[(long)m * N + n] = v;
            }
        }
}

// ---------------------------------------------------------------------------
// logits[b] = h_all[b, idx_b*256 .. +256] @ Wc2[idx_b] + bc2[idx_b]
// one wave per sample, shuffle reduction
// ---------------------------------------------------------------------------
__global__ __launch_bounds__(256)
void k_head_gather(const float* __restrict__ h_all, const int* __restrict__ idx,
                   const float* __restrict__ Wc2, const float* __restrict__ bc2,
                   float* __restrict__ logits) {
    const int wave = threadIdx.x >> 5, lane = threadIdx.x & 31;
    const int b = blockIdx.x * 8 + wave;
    const int d = idx[b];
    const float* h = h_all + (long)b * (ND * 256) + d * 256;
    const float* W = Wc2 + (long)d * 256 * NC;
    float a0 = 0.0f, a1 = 0.0f;
    for (int k = lane; k < 256; k += 32) {
        const float v = h[k];
        a0 += v * W[k * NC + 0];
        a1 += v * W[k * NC + 1];
    }
#pragma unroll
    for (int off = 16; off >= 1; off >>= 1) {
        a0 += __shfl_down(a0, off, 32);
        a1 += __shfl_down(a1, off, 32);
    }
    if (lane == 0) {
        logits[b * NC + 0] = a0 + bc2[d * NC + 0];
        logits[b * NC + 1] = a1 + bc2[d * NC + 1];
    }
}

// ---------------------------------------------------------------------------
// emotion[b] = e1[b] (128) @ We2 (128x8) + be2 ; one wave per sample
// ---------------------------------------------------------------------------
__global__ __launch_bounds__(256)
void k_emotion(const float* __restrict__ e1, const float* __restrict__ We2,
               const float* __restrict__ be2, float* __restrict__ out) {
    const int wave = threadIdx.x >> 5, lane = threadIdx.x & 31;
    const int b = blockIdx.x * 8 + wave;
    float acc[NE] = {};
    for (int k = lane; k < 128; k += 32) {
        const float v = e1[(long)b * 128 + k];
        const float* w = We2 + k * NE;
#pragma unroll
        for (int e = 0; e < NE; ++e) acc[e] += v * w[e];
    }
#pragma unroll
    for (int off = 16; off >= 1; off >>= 1)
#pragma unroll
        for (int e = 0; e < NE; ++e) acc[e] += __shfl_down(acc[e], off, 32);
    if (lane == 0) {
#pragma unroll
        for (int e = 0; e < NE; ++e) out[b * NE + e] = acc[e] + be2[e];
    }
}

// ---------------------------------------------------------------------------
extern "C" void kernel_launch(void* const* d_in, const int* in_sizes, int n_in,
                              void* d_out, int out_size, void* d_ws, size_t ws_size,
                              hipStream_t stream) {
    (void)in_sizes; (void)n_in; (void)out_size; (void)ws_size;
    const float* audio = (const float*)d_in[0];
    const float* video = (const float*)d_in[1];
    const int*   didx  = (const int*)d_in[2];
    const float* Wa    = (const float*)d_in[3];
    const float* ba    = (const float*)d_in[4];
    const float* Wv    = (const float*)d_in[5];
    const float* bv    = (const float*)d_in[6];
    const float* Pa    = (const float*)d_in[7];
    const float* Cva   = (const float*)d_in[8];
    const float* Pv    = (const float*)d_in[9];
    const float* Cav   = (const float*)d_in[10];
    const float* Wproj = (const float*)d_in[11];
    const float* bproj = (const float*)d_in[12];
    const float* Wf    = (const float*)d_in[13];
    const float* bfv   = (const float*)d_in[14];
    const float* Wc1   = (const float*)d_in[15];
    const float* bc1   = (const float*)d_in[16];
    const float* Wc2   = (const float*)d_in[17];
    const float* bc2   = (const float*)d_in[18];
    const float* We1   = (const float*)d_in[19];
    const float* be1   = (const float*)d_in[20];
    const float* We2   = (const float*)d_in[21];
    const float* be2   = (const float*)d_in[22];

    float* outp = (float*)d_out;
    // output tuple: (logits, af, vf, fused, emotion)
    float* o_logits  = outp;
    float* o_af      = o_logits + (long)BATCH * NC;
    float* o_vf      = o_af + (long)BATCH * DIMH;
    float* o_fused   = o_vf + (long)BATCH * DIMH;
    float* o_emotion = o_fused + (long)BATCH * DIMF;

    // ---- workspace carve-out ----
    char* p = (char*)d_ws;
    auto take = [&](size_t bytes) { void* r = (void*)p; p += (bytes + 255) & ~(size_t)255; return r; };
    __bf16* WTa   = (__bf16*)take((size_t)DIMH * DIMA * 2);        // 512x768
    __bf16* WTv   = (__bf16*)take((size_t)DIMH * DIMV * 2);        // 512x2048
    __bf16* WTPa  = (__bf16*)take((size_t)DIMF * DIMH * 2);        // 1024x512
    __bf16* WTCva = (__bf16*)take((size_t)DIMF * DIMH * 2);
    __bf16* WTPv  = (__bf16*)take((size_t)DIMF * DIMH * 2);
    __bf16* WTCav = (__bf16*)take((size_t)DIMF * DIMH * 2);
    __bf16* WTpT  = (__bf16*)take((size_t)DIMF * DIMF * 2);        // proj top   1024x1024
    __bf16* WTpB  = (__bf16*)take((size_t)DIMF * DIMF * 2);        // proj bottom
    __bf16* WTf0  = (__bf16*)take((size_t)DIMF * DIMF * 2);
    __bf16* WTf1  = (__bf16*)take((size_t)DIMF * DIMF * 2);
    __bf16* WTf2  = (__bf16*)take((size_t)DIMF * DIMF * 2);
    __bf16* WTe1  = (__bf16*)take((size_t)128 * DIMH * 2);         // 128x512
    __bf16* WTc1  = (__bf16*)take((size_t)ND * 256 * DIMF * 2);    // 4096x1024
    // activation pool (4 buffers, contiguous; reused as h_all 4096x4096 later)
    float* act = (float*)take((size_t)4 * BATCH * DIMF * 4);
    float* aa  = act;
    float* av  = act + (long)BATCH * DIMF;
    float* fA  = act + 2L * BATCH * DIMF;
    float* fB  = act + 3L * BATCH * DIMF;
    float* h_all = act;                                            // 4096x4096 alias
    float* e1  = (float*)take((size_t)BATCH * 128 * 4);

    const dim3 blk(256, 1, 1);

    // ---- 1) one-time weight transposes fp32 -> bf16 N-major ----
    hipLaunchKernelGGL(k_transpose_bf16, dim3(DIMH/32, DIMA/32, 1), blk, 0, stream, Wa,  WTa,  DIMA, DIMH, 0, 0);
    hipLaunchKernelGGL(k_transpose_bf16, dim3(DIMH/32, DIMV/32, 1), blk, 0, stream, Wv,  WTv,  DIMV, DIMH, 0, 0);
    hipLaunchKernelGGL(k_transpose_bf16, dim3(DIMF/32, DIMH/32, 1), blk, 0, stream, Pa,  WTPa, DIMH, DIMF, 0, 0);
    hipLaunchKernelGGL(k_transpose_bf16, dim3(DIMF/32, DIMH/32, 1), blk, 0, stream, Cva, WTCva,DIMH, DIMF, 0, 0);
    hipLaunchKernelGGL(k_transpose_bf16, dim3(DIMF/32, DIMH/32, 1), blk, 0, stream, Pv,  WTPv, DIMH, DIMF, 0, 0);
    hipLaunchKernelGGL(k_transpose_bf16, dim3(DIMF/32, DIMH/32, 1), blk, 0, stream, Cav, WTCav,DIMH, DIMF, 0, 0);
    hipLaunchKernelGGL(k_transpose_bf16, dim3(DIMF/32, DIMF/32, 1), blk, 0, stream, Wproj,                 WTpT, DIMF, DIMF, 0, 0);
    hipLaunchKernelGGL(k_transpose_bf16, dim3(DIMF/32, DIMF/32, 1), blk, 0, stream, Wproj + (long)DIMF*DIMF, WTpB, DIMF, DIMF, 0, 0);
    hipLaunchKernelGGL(k_transpose_bf16, dim3(DIMF/32, DIMF/32, 1), blk, 0, stream, Wf,                    WTf0, DIMF, DIMF, 0, 0);
    hipLaunchKernelGGL(k_transpose_bf16, dim3(DIMF/32, DIMF/32, 1), blk, 0, stream, Wf + 1L*DIMF*DIMF,     WTf1, DIMF, DIMF, 0, 0);
    hipLaunchKernelGGL(k_transpose_bf16, dim3(DIMF/32, DIMF/32, 1), blk, 0, stream, Wf + 2L*DIMF*DIMF,     WTf2, DIMF, DIMF, 0, 0);
    hipLaunchKernelGGL(k_transpose_bf16, dim3(128/32, DIMH/32, 1),  blk, 0, stream, We1, WTe1, DIMH, 128, 0, 0);
    hipLaunchKernelGGL(k_transpose_bf16, dim3(256/32, DIMF/32, ND), blk, 0, stream, Wc1, WTc1, DIMF, 256,
                       (long)DIMF * 256, (long)256 * DIMF);

    const int MB = BATCH / BM; // 64
    // ---- 2) encoders ----
    hipLaunchKernelGGL(k_gemm_bf16, dim3(MB, DIMH/BN, 1), blk, 0, stream,
                       audio, WTa, DIMA, nullptr, nullptr, 0, ba, o_af, DIMH, 1);
    hipLaunchKernelGGL(k_gemm_bf16, dim3(MB, DIMH/BN, 1), blk, 0, stream,
                       video, WTv, DIMV, nullptr, nullptr, 0, bv, o_vf, DIMH, 1);
    // ---- 3) cross-modal mixing (dual-pair fused GEMMs) ----
    hipLaunchKernelGGL(k_gemm_bf16, dim3(MB, DIMF/BN, 1), blk, 0, stream,
                       o_af, WTPa, DIMH, o_vf, WTCva, DIMH, nullptr, aa, DIMF, 0);
    hipLaunchKernelGGL(k_gemm_bf16, dim3(MB, DIMF/BN, 1), blk, 0, stream,
                       o_vf, WTPv, DIMH, o_af, WTCav, DIMH, nullptr, av, DIMF, 0);
    // ---- 4) concat projection: aa@Wtop + av@Wbot + bproj ----
    hipLaunchKernelGGL(k_gemm_bf16, dim3(MB, DIMF/BN, 1), blk, 0, stream,
                       aa, WTpT, DIMF, av, WTpB, DIMF, bproj, fA, DIMF, 0);
    // ---- 5) fusion MLP (3 layers) ----
    hipLaunchKernelGGL(k_gemm_bf16, dim3(MB, DIMF/BN, 1), blk, 0, stream,
                       fA, WTf0, DIMF, nullptr, nullptr, 0, bfv + 0L*DIMF, fB, DIMF, 1);
    hipLaunchKernelGGL(k_gemm_bf16, dim3(MB, DIMF/BN, 1), blk, 0, stream,
                       fB, WTf1, DIMF, nullptr, nullptr, 0, bfv + 1L*DIMF, fA, DIMF, 1);
    hipLaunchKernelGGL(k_gemm_bf16, dim3(MB, DIMF/BN, 1), blk, 0, stream,
                       fA, WTf2, DIMF, nullptr, nullptr, 0, bfv + 2L*DIMF, o_fused, DIMF, 1);
    // ---- 6) emotion head stage 1 ----
    hipLaunchKernelGGL(k_gemm_bf16, dim3(MB, 128/BN, 1), blk, 0, stream,
                       o_vf, WTe1, DIMH, nullptr, nullptr, 0, be1, e1, 128, 1);
    // ---- 7) dense all-head classifier: h_all = relu(fused @ Wc1_all + bc1) ----
    hipLaunchKernelGGL(k_gemm_bf16, dim3(MB, (ND*256)/BN, 1), blk, 0, stream,
                       o_fused, WTc1, DIMF, nullptr, nullptr, 0, bc1, h_all, ND*256, 1);
    // ---- 8) gather + second stages ----
    hipLaunchKernelGGL(k_head_gather, dim3(BATCH/8, 1, 1), blk, 0, stream,
                       h_all, didx, Wc2, bc2, o_logits);
    hipLaunchKernelGGL(k_emotion, dim3(BATCH/8, 1, 1), blk, 0, stream,
                       e1, We2, be2, o_emotion);
}
